// AttentionLayer_86835648791168
// MI455X (gfx1250) — matrix operand order repaired and checked by reading, not compiled, not run on hardware
//
#include <hip/hip_runtime.h>
#include <hip/hip_bf16.h>
#include <stdint.h>

// ---------------------------------------------------------------------------
// CDNA5 (gfx1250, wave32) attention pipeline:
//   fp32 -> bf16 once up front (weights pre-transposed), every GEMM consumes
//   B in [N,K] layout so ALL LDS staging is contiguous async-to-LDS b128,
//   double-buffered, feeding v_wmma_f32_16x16x32_bf16.
// ---------------------------------------------------------------------------

typedef __attribute__((ext_vector_type(16))) __bf16 bf16x16;
typedef __attribute__((ext_vector_type(8)))  float  f32x8;
typedef __attribute__((ext_vector_type(4)))  int    v4i;

#define TB     256   // threads per block = 8 wave32s
#define BM     128
#define BN     128
#define BK     32    // matches WMMA K for bf16
#define LDSPAD 40    // padded row length (ushorts) -> 80B row stride

#if defined(__HIP_DEVICE_COMPILE__) && \
    __has_builtin(__builtin_amdgcn_global_load_async_to_lds_b128) && \
    __has_builtin(__builtin_amdgcn_s_wait_asynccnt)
#define HAVE_ASYNC 1
#else
#define HAVE_ASYNC 0
#endif

__device__ __forceinline__ unsigned short f2bf(float f) {
  // round-to-nearest-even fp32 -> bf16
  unsigned int u = __float_as_uint(f);
  u += 0x7FFFu + ((u >> 16) & 1u);
  return (unsigned short)(u >> 16);
}
__device__ __forceinline__ unsigned int pk2(float a, float b) {
  return (unsigned int)f2bf(a) | ((unsigned int)f2bf(b) << 16);
}

// Copy 8 bf16 (16 bytes) global -> LDS. Async (ASYNCcnt-tracked) when available.
__device__ __forceinline__ void cp16(unsigned short* lds, const unsigned short* g) {
#if HAVE_ASYNC
  __builtin_amdgcn_global_load_async_to_lds_b128(
      (__attribute__((address_space(1))) v4i*)(unsigned short*)g,
      (__attribute__((address_space(3))) v4i*)lds, 0, 0);
#else
  *(uint4*)lds = *(const uint4*)g;
#endif
}
__device__ __forceinline__ void wait_async_copies() {
#if HAVE_ASYNC
  __builtin_amdgcn_s_wait_asynccnt(0);
#endif
}

union FragCast {
  uint4   q[2];   // two ds_load_b128 payloads
  bf16x16 v;      // 8 VGPRs of packed bf16 for the WMMA operand
};

// C[M,N] = scale * (A x Bt^T):
//   A  : bf16 [M,K] row-major
//   Bt : bf16 [N,K] row-major (i.e. B transposed -> contiguous K per column)
//   OUTBF16=1: C stored bf16, else fp32.
template <bool OUTBF16>
__global__ __launch_bounds__(TB)
void gemm_bt_wmma(const unsigned short* __restrict__ A,
                  const unsigned short* __restrict__ Bt,
                  void* __restrict__ Cv,
                  int M, int N, int K,
                  long long sA, long long sB, long long sC, float scale)
{
  __shared__ __align__(16) unsigned short As[2][BM * LDSPAD];
  __shared__ __align__(16) unsigned short Bs[2][BN * LDSPAD];

  const int tid  = threadIdx.x;
  const int lane = tid & 31;
  const int h    = lane >> 4;   // lane half (K sub-range select per ISA layout)
  const int ln   = lane & 15;   // row (A) / column (B,C) within 16x16 tile
  const int wave = tid >> 5;
  const int wm   = (wave >> 2) << 6;  // wave row offset: 0 / 64
  const int wn   = (wave &  3) << 5;  // wave col offset: 0..96

  const int brow = blockIdx.y * BM;
  const int bcol = blockIdx.x * BN;

  const unsigned short* Ab = A  + (long long)blockIdx.z * sA;
  const unsigned short* Bb = Bt + (long long)blockIdx.z * sB;

  const int am  = tid >> 1;        // 0..127: staged row
  const int ak0 = (tid & 1) << 4;  // 0 or 16: staged k-offset

  auto stageA = [&](int kb, unsigned short* dst) {
    const unsigned short* src = Ab + (long long)(brow + am) * K + (kb + ak0);
    cp16(&dst[am * LDSPAD + ak0], src);
    cp16(&dst[am * LDSPAD + ak0 + 8], src + 8);
  };
  auto stageB = [&](int kb, unsigned short* dst) {
    const unsigned short* src = Bb + (long long)(bcol + am) * K + (kb + ak0);
    cp16(&dst[am * LDSPAD + ak0], src);
    cp16(&dst[am * LDSPAD + ak0 + 8], src + 8);
  };

  f32x8 acc[4][2];
#pragma unroll
  for (int i = 0; i < 4; ++i)
#pragma unroll
    for (int j = 0; j < 2; ++j)
#pragma unroll
      for (int e = 0; e < 8; ++e)
        acc[i][j][e] = 0.0f;

  // prologue: fill buffer 0
  stageA(0, As[0]);
  stageB(0, Bs[0]);
  wait_async_copies();
  __syncthreads();

  int cur = 0;
  for (int kb = 0; kb < K; kb += BK) {
    const int nxt = cur ^ 1;
    if (kb + BK < K) {          // prefetch next slab into the other buffer
      stageA(kb + BK, As[nxt]);
      stageB(kb + BK, Bs[nxt]);
    }

    // ---- fragments per CDNA5 16-bit WMMA VGPR layouts --------------------
    // A (16x32): lanes 0-15 -> K {0..7,16..23}; lanes 16-31 -> K {8..15,24..31}
    // B (32x16): lanes 0-15 -> K {0..7,16..23}; lanes 16-31 -> K {8..15,24..31}
    FragCast aF[4], bF[2];
#pragma unroll
    for (int i = 0; i < 4; ++i) {
      const unsigned short* base = &As[cur][(wm + i * 16 + ln) * LDSPAD];
      aF[i].q[0] = *(const uint4*)&base[h * 8];
      aF[i].q[1] = *(const uint4*)&base[16 + h * 8];
    }
#pragma unroll
    for (int j = 0; j < 2; ++j) {
      const unsigned short* base = &Bs[cur][(wn + j * 16 + ln) * LDSPAD];
      bF[j].q[0] = *(const uint4*)&base[h * 8];
      bF[j].q[1] = *(const uint4*)&base[16 + h * 8];
    }
#pragma unroll
    for (int i = 0; i < 4; ++i)
#pragma unroll
      for (int j = 0; j < 2; ++j)
        acc[i][j] = __builtin_amdgcn_wmma_f32_16x16x32_bf16(
            false, aF[i].v, false, bF[j].v, (short)0, acc[i][j], false, false);

    wait_async_copies();
    __syncthreads();
    cur = nxt;
  }

  // ---- epilogue: C layout VGPR e -> M = e + 8*h, N = ln --------------------
#pragma unroll
  for (int i = 0; i < 4; ++i)
#pragma unroll
    for (int j = 0; j < 2; ++j) {
      const int col = bcol + wn + j * 16 + ln;
#pragma unroll
      for (int e = 0; e < 8; ++e) {
        const int row = brow + wm + i * 16 + h * 8 + e;
        const float val = acc[i][j][e] * scale;
        if constexpr (OUTBF16) {
          unsigned short* Cb = (unsigned short*)Cv + (long long)blockIdx.z * sC;
          Cb[(long long)row * N + col] = f2bf(val);
        } else {
          float* Cb = (float*)Cv + (long long)blockIdx.z * sC;
          Cb[(long long)row * N + col] = val;
        }
      }
    }
}

// fp32 -> bf16 bulk conversion (n must be a multiple of 8; true for all uses)
__global__ __launch_bounds__(TB)
void f32_to_bf16(const float* __restrict__ src, unsigned short* __restrict__ dst,
                 long long n)
{
  const long long i = ((long long)blockIdx.x * TB + threadIdx.x) * 8;
  if (i >= n) return;
  float4 a = *(const float4*)(src + i);
  float4 b = *(const float4*)(src + i + 4);
  uint4 p;
  p.x = pk2(a.x, a.y); p.y = pk2(a.z, a.w);
  p.z = pk2(b.x, b.y); p.w = pk2(b.z, b.w);
  *(uint4*)(dst + i) = p;
}

// fp32 [n,n] -> bf16 transposed [n,n] (LDS-tiled, coalesced both sides)
__global__ __launch_bounds__(TB)
void transpose_f32_to_bf16(const float* __restrict__ src,
                           unsigned short* __restrict__ dst, int n)
{
  __shared__ unsigned short tile[32][33];
  const int tr = threadIdx.x >> 5;   // 0..7
  const int tc = threadIdx.x & 31;   // 0..31
  const int r0 = blockIdx.y * 32;
  const int c0 = blockIdx.x * 32;
#pragma unroll
  for (int i = 0; i < 4; ++i)
    tile[i * 8 + tr][tc] = f2bf(src[(long long)(r0 + i * 8 + tr) * n + (c0 + tc)]);
  __syncthreads();
#pragma unroll
  for (int i = 0; i < 4; ++i)
    dst[(long long)(c0 + i * 8 + tr) * n + (r0 + tc)] = tile[tc][i * 8 + tr];
}

// Row softmax over rows of length 2048; writes fp32 result (required output)
// and a bf16 shadow copy used by the weights @ V GEMM.
__global__ __launch_bounds__(TB)
void softmax2048(float* __restrict__ W, unsigned short* __restrict__ Wb)
{
  const long long base = (long long)blockIdx.x * 2048;
  float* p = W + base;
  unsigned short* pb = Wb + base;
  const int t = threadIdx.x;
  __shared__ float red[TB];

  float v[8];
#pragma unroll
  for (int j = 0; j < 8; ++j) v[j] = p[t + j * 256];

  float m = v[0];
#pragma unroll
  for (int j = 1; j < 8; ++j) m = fmaxf(m, v[j]);
  red[t] = m; __syncthreads();
  for (int s = 128; s > 0; s >>= 1) {
    if (t < s) red[t] = fmaxf(red[t], red[t + s]);
    __syncthreads();
  }
  m = red[0]; __syncthreads();

  float sum = 0.0f;
#pragma unroll
  for (int j = 0; j < 8; ++j) { v[j] = __expf(v[j] - m); sum += v[j]; }
  red[t] = sum; __syncthreads();
  for (int s = 128; s > 0; s >>= 1) {
    if (t < s) red[t] += red[t + s];
    __syncthreads();
  }
  const float inv = 1.0f / red[0];
#pragma unroll
  for (int j = 0; j < 8; ++j) {
    const float w = v[j] * inv;
    p[t + j * 256]  = w;
    pb[t + j * 256] = f2bf(w);
  }
}

// ---------------------------------------------------------------------------
extern "C" void kernel_launch(void* const* d_in, const int* in_sizes, int n_in,
                              void* d_out, int out_size, void* d_ws, size_t ws_size,
                              hipStream_t stream)
{
  (void)in_sizes; (void)n_in; (void)out_size; (void)ws_size;

  const int Bt = 8, S = 2048, D = 1024;
  const long long SD  = (long long)S * D;        // 2,097,152
  const long long SS  = (long long)S * S;        // 4,194,304
  const long long BSD = (long long)Bt * SD;      // 16,777,216
  const long long BSS = (long long)Bt * SS;      // 33,554,432
  const long long DD  = (long long)D * D;        // 1,048,576

  const float* input   = (const float*)d_in[0];
  const float* context = (const float*)d_in[1];
  const float* Wq      = (const float*)d_in[2];
  const float* Wk      = (const float*)d_in[3];
  const float* Wv      = (const float*)d_in[4];
  const float* Wo      = (const float*)d_in[5];

  float* out     = (float*)d_out;                 // [B,S,D] fp32
  float* weights = out + BSD;                     // [B,S,S] fp32

  // workspace: everything bf16 (unsigned short raw)
  unsigned short* inB = (unsigned short*)d_ws;    // input   [B,S,D]
  unsigned short* cxB = inB + BSD;                // context [B,S,D]
  unsigned short* WqT = cxB + BSD;                // Wq^T [D,D]
  unsigned short* WkT = WqT + DD;                 // Wk^T [D,D]
  unsigned short* WvT = WkT + DD;                 // Wv^T [D,D]
  unsigned short* WoT = WvT + DD;                 // Wo^T [D,D]
  unsigned short* QB  = WoT + DD;                 // Q   [B,S,D]
  unsigned short* KB  = QB + BSD;                 // K   [B,S,D]  (= Bt for scores)
  unsigned short* VT  = KB + BSD;                 // V^T [B,D,S]  (= Bt for ctx)
  unsigned short* wB  = VT + BSD;                 // softmax weights bf16 [B,S,S]
  unsigned short* cB  = wB + BSS;                 // attention context [B,S,D]

  dim3 blk(TB);
  dim3 gSD(D / BN, S / BM, Bt);   // (8,16,8):  M=S, N=D
  dim3 gSS(S / BN, S / BM, Bt);   // (16,16,8): M=S, N=S
  dim3 gDS(S / BN, D / BM, Bt);   // (16, 8,8): M=D, N=S
  dim3 gTr(D / 32, D / 32);       // (32,32)

  // one-time conversions: activations straight, weights transposed
  f32_to_bf16<<<dim3((unsigned)(BSD / (8 * TB))), blk, 0, stream>>>(input,   inB, BSD);
  f32_to_bf16<<<dim3((unsigned)(BSD / (8 * TB))), blk, 0, stream>>>(context, cxB, BSD);
  transpose_f32_to_bf16<<<gTr, blk, 0, stream>>>(Wq, WqT, D);
  transpose_f32_to_bf16<<<gTr, blk, 0, stream>>>(Wk, WkT, D);
  transpose_f32_to_bf16<<<gTr, blk, 0, stream>>>(Wv, WvT, D);
  transpose_f32_to_bf16<<<gTr, blk, 0, stream>>>(Wo, WoT, D);

  // Q = input @ Wq ; K = context @ Wk   (Bt = W^T)
  gemm_bt_wmma<true><<<gSD, blk, 0, stream>>>(inB, WqT, QB, S, D, D, SD, 0, SD, 1.0f);
  gemm_bt_wmma<true><<<gSD, blk, 0, stream>>>(cxB, WkT, KB, S, D, D, SD, 0, SD, 1.0f);

  // V^T = Wv^T @ context^T : A = Wv^T [D,D], Bt = context [S,D] -> C = V^T [D,S]
  gemm_bt_wmma<true><<<gDS, blk, 0, stream>>>(WvT, cxB, VT, D, S, D, 0, SD, SD, 1.0f);

  // scores = (Q @ K^T) / sqrt(D): Bt = K [S,D] -> fp32 weights region of d_out
  gemm_bt_wmma<false><<<gSS, blk, 0, stream>>>(QB, KB, weights, S, S, D, SD, SD, SS, 0.03125f);

  // softmax rows (fp32 out + bf16 shadow)
  softmax2048<<<dim3((unsigned)(Bt * S)), blk, 0, stream>>>(weights, wB);

  // ctx = weights @ V : Bt = V^T [D,S] ;  out = ctx @ Wo : Bt = Wo^T
  gemm_bt_wmma<true ><<<gSD, blk, 0, stream>>>(wB, VT,  cB,  S, D, S, SS, SD, SD, 1.0f);
  gemm_bt_wmma<false><<<gSD, blk, 0, stream>>>(cB, WoT, out, S, D, D, SD, 0,  SD, 1.0f);
}
